// RVQ_33097017983693
// MI455X (gfx1250) — matrix operand order repaired and checked by reading, not compiled, not run on hardware
//
#include <hip/hip_runtime.h>
#include <hip/hip_bf16.h>

// ---------------------------------------------------------------------------
// Multi-stage residual VQ for MI455X (gfx1250, wave32, WMMA).
//   N=65536 rows, D=256 dims, S=6 stages, K=1024 codewords/stage.
// v_wmma_f32_16x16x32_bf16 distances; 8 waves/WG share each codebook tile,
// staged double-buffered in LDS via global_load_async_to_lds_b128 (ASYNCcnt),
// cutting L2 codebook traffic 8x. ||c||^2 is also async-staged to LDS, so the
// steady-state loop contains no VMEM loads at all (async DMA + DS + WMMA).
// f32 residual lives in a per-lane swizzled LDS slab.
// out = input - final_residual  (== sum of per-stage quantized vectors)
// ---------------------------------------------------------------------------

typedef __attribute__((ext_vector_type(16))) __bf16 v16bf;
typedef __attribute__((ext_vector_type(8)))  __bf16 v8bf;
typedef __attribute__((ext_vector_type(8)))  float  v8f;

#define DDIM   256
#define KCB    1024
#define NSTAGE 6
#define NROWS  65536
#define TROW   528            // padded B-tile row pitch in bytes (512 + 16)
#define TSZ    (16 * TROW)    // one B tile in LDS: 8448 B

// ---------------------------------------------------------------------------
// Prep: codebooks f32 -> bf16, and per-codeword squared norms.
// ---------------------------------------------------------------------------
__global__ __launch_bounds__(64) void rvq_prep(const float* __restrict__ cb,
                                               __bf16* __restrict__ cb16,
                                               float* __restrict__ cnorm) {
    const int cw = blockIdx.x;          // 0 .. S*K-1
    const int t  = threadIdx.x;         // 0 .. 63, 4 dims each
    const float4 v = *(const float4*)(cb + (size_t)cw * DDIM + t * 4);
    __bf16* dst = cb16 + (size_t)cw * DDIM + t * 4;
    dst[0] = (__bf16)v.x; dst[1] = (__bf16)v.y;
    dst[2] = (__bf16)v.z; dst[3] = (__bf16)v.w;

    float ss = v.x * v.x + v.y * v.y + v.z * v.z + v.w * v.w;
    #pragma unroll
    for (int off = 16; off > 0; off >>= 1)
        ss += __shfl_down(ss, off, 32);

    __shared__ float partial[2];
    if ((t & 31) == 0) partial[t >> 5] = ss;
    __syncthreads();
    if (t == 0) cnorm[cw] = partial[0] + partial[1];
}

// ---------------------------------------------------------------------------
// Main kernel: 256 threads = 8 waves; each wave owns 16 rows (128 rows/WG).
// ---------------------------------------------------------------------------
__global__ __launch_bounds__(256) void rvq_main(const float* __restrict__ x,
                                                const float* __restrict__ cbf,
                                                const __bf16* __restrict__ cb16,
                                                const float* __restrict__ cnorm,
                                                float* __restrict__ out) {
    // Residual slab: 128 floats/lane as 32 float4 at stride 32 float4s.
    __shared__ float4 rbuf[8 * 1024];                      // 128 KB
    // Double-buffered shared B tile (async-DMA target), 528B row pitch.
    __shared__ __align__(16) unsigned char btile[2 * TSZ]; // 16.5 KB
    // Per-stage ||c||^2, async-staged once per stage.
    __shared__ __align__(16) float cnlds[KCB];             // 4 KB

    const int t    = threadIdx.x;
    const int lane = t & 31;
    const int wave = t >> 5;
    const int half = lane >> 4;         // K-range half (A/B layouts)
    const int nloc = lane & 15;         // row (A) / column-in-tile (B, C)
    const int row  = (blockIdx.x * 8 + wave) * 16 + nloc;

    float4* const myr = rbuf + wave * 1024 + lane;         // myr[i*32], i=0..31
    const float* xrow = x + (size_t)row * DDIM;

    // Cooperative staging map: thread t moves 32 B of the 8 KB tile.
    const int srow = t >> 4;                 // tile row 0..15
    const int scol = (t & 15) * 32;          // byte offset within row
    const unsigned lds_st = (unsigned)(uintptr_t)btile + srow * TROW + scol;
    const unsigned lds_cn = (unsigned)(uintptr_t)cnlds + t * 16;

    // Issue 32B (2 x b128) of tile TILE into LDS buffer BUFSEL (ASYNCcnt).
    #define ASYNC_STAGE(BUFSEL, TILE)                                         \
    {                                                                         \
        const unsigned _l = lds_st + (BUFSEL) * TSZ;                          \
        const unsigned char* _g = gstage + (size_t)(TILE) * 8192 + t * 32;    \
        asm volatile("global_load_async_to_lds_b128 %0, %1, off\n\t"          \
                     "global_load_async_to_lds_b128 %0, %1, off offset:16"    \
                     :: "v"(_l), "v"(_g) : "memory");                         \
    }
    // Stage this stage's 4 KB ||c||^2 vector: 16 B per thread.
    #define ASYNC_CNORM()                                                     \
    {                                                                         \
        const unsigned char* _g = (const unsigned char*)cns + t * 16;         \
        asm volatile("global_load_async_to_lds_b128 %0, %1, off"              \
                     :: "v"(lds_cn), "v"(_g) : "memory");                     \
    }
    #define WAIT_ASYNC() asm volatile("s_wait_asynccnt 0" ::: "memory")

    // ---- load input into the residual slab -------------------------------
    // slot i=4*dc+{0,1} <-> d in [dc*32+half*8, +8); i=4*dc+{2,3} <-> +16.
    #pragma unroll
    for (int dc = 0; dc < 8; ++dc) {
        const int d0 = dc * 32 + half * 8;
        myr[(4 * dc + 0) * 32] = *(const float4*)(xrow + d0);
        myr[(4 * dc + 1) * 32] = *(const float4*)(xrow + d0 + 4);
        myr[(4 * dc + 2) * 32] = *(const float4*)(xrow + d0 + 16);
        myr[(4 * dc + 3) * 32] = *(const float4*)(xrow + d0 + 20);
    }

    for (int s = 0; s < NSTAGE; ++s) {
        const unsigned char* gstage =
            (const unsigned char*)(cb16 + (size_t)s * KCB * DDIM);
        const float* cns = cnorm + s * KCB;

        // ---- prologue: async-stage tile 0 and ||c||^2 --------------------
        ASYNC_STAGE(0, 0);
        ASYNC_CNORM();

        // ---- build bf16 A fragments from the f32 residual ----------------
        v16bf A[8];
        #pragma unroll
        for (int dc = 0; dc < 8; ++dc) {
            const float4 q0 = myr[(4 * dc + 0) * 32];
            const float4 q1 = myr[(4 * dc + 1) * 32];
            const float4 q2 = myr[(4 * dc + 2) * 32];
            const float4 q3 = myr[(4 * dc + 3) * 32];
            v16bf a;
            a[0]  = (__bf16)q0.x; a[1]  = (__bf16)q0.y;
            a[2]  = (__bf16)q0.z; a[3]  = (__bf16)q0.w;
            a[4]  = (__bf16)q1.x; a[5]  = (__bf16)q1.y;
            a[6]  = (__bf16)q1.z; a[7]  = (__bf16)q1.w;
            a[8]  = (__bf16)q2.x; a[9]  = (__bf16)q2.y;
            a[10] = (__bf16)q2.z; a[11] = (__bf16)q2.w;
            a[12] = (__bf16)q3.x; a[13] = (__bf16)q3.y;
            a[14] = (__bf16)q3.z; a[15] = (__bf16)q3.w;
            A[dc] = a;
        }

        float bestv[8];
        int   bestk[8];
        #pragma unroll
        for (int j = 0; j < 8; ++j) { bestv[j] = 3.402823466e+38f; bestk[j] = 0; }

        WAIT_ASYNC();
        __syncthreads();

        for (int ct = 0; ct < 64; ++ct) {
            if (ct < 63) ASYNC_STAGE((ct + 1) & 1, ct + 1);   // next tile -> other buf

            // ||c||^2 for this tile: conflict-free broadcast DS read.
            const float cn = cnlds[ct * 16 + nloc];

            // Fragment reads from the shared tile: lane = column nloc,
            // two 16 B chunks (k = 8*half.., 16+8*half..) per dc at conflict-free pitch.
            const unsigned char* bt =
                btile + (ct & 1) * TSZ + nloc * TROW + half * 16;
            v8f acc = {};
            #pragma unroll
            for (int dc = 0; dc < 8; ++dc) {
                const v8bf b0 = *(const v8bf*)(bt + dc * 64);
                const v8bf b1 = *(const v8bf*)(bt + dc * 64 + 32);
                const v16bf B = __builtin_shufflevector(b0, b1,
                        0, 1, 2, 3, 4, 5, 6, 7, 8, 9, 10, 11, 12, 13, 14, 15);
                acc = __builtin_amdgcn_wmma_f32_16x16x32_bf16(
                          false, A[dc], false, B, (short)0, acc, false, false);
            }

            const int k = ct * 16 + nloc;
            #pragma unroll
            for (int j = 0; j < 8; ++j) {
                const float dist = cn - 2.0f * acc[j];     // ||r||^2 omitted
                if (dist < bestv[j]) { bestv[j] = dist; bestk[j] = k; }
            }

            WAIT_ASYNC();        // own async stores to the other buffer done
            __syncthreads();     // tile visible to all waves before next read
        }

        // ---- argmin across the 16 lanes of each half; row m = half*8+j ---
        int redk[8];
        #pragma unroll
        for (int j = 0; j < 8; ++j) {
            float v = bestv[j];
            int   kk = bestk[j];
            #pragma unroll
            for (int off = 1; off < 16; off <<= 1) {
                const float ov = __shfl_xor(v, off, 32);
                const int   ok = __shfl_xor(kk, off, 32);
                if (ov < v || (ov == v && ok < kk)) { v = ov; kk = ok; }
            }
            redk[j] = kk;   // valid in every lane of this half
        }

        // ---- broadcast: this lane needs the winner of row `nloc` ---------
        const int jj = nloc & 7;
        int mine = redk[0];
        #pragma unroll
        for (int j = 1; j < 8; ++j) mine = (jj == j) ? redk[j] : mine;
        const int other = __shfl_xor(mine, 16, 32);
        const int kk = ((nloc >> 3) == half) ? mine : other;

        // ---- residual update in f32: r -= codebook[kk] -------------------
        const float* crow = cbf + ((size_t)s * KCB + kk) * DDIM;
        #pragma unroll
        for (int dc = 0; dc < 8; ++dc) {
            const int d0 = dc * 32 + half * 8;
            const float4 c0 = *(const float4*)(crow + d0);
            const float4 c1 = *(const float4*)(crow + d0 + 4);
            const float4 c2 = *(const float4*)(crow + d0 + 16);
            const float4 c3 = *(const float4*)(crow + d0 + 20);
            float4 r0 = myr[(4 * dc + 0) * 32];
            float4 r1 = myr[(4 * dc + 1) * 32];
            float4 r2 = myr[(4 * dc + 2) * 32];
            float4 r3 = myr[(4 * dc + 3) * 32];
            r0.x -= c0.x; r0.y -= c0.y; r0.z -= c0.z; r0.w -= c0.w;
            r1.x -= c1.x; r1.y -= c1.y; r1.z -= c1.z; r1.w -= c1.w;
            r2.x -= c2.x; r2.y -= c2.y; r2.z -= c2.z; r2.w -= c2.w;
            r3.x -= c3.x; r3.y -= c3.y; r3.z -= c3.z; r3.w -= c3.w;
            myr[(4 * dc + 0) * 32] = r0;
            myr[(4 * dc + 1) * 32] = r1;
            myr[(4 * dc + 2) * 32] = r2;
            myr[(4 * dc + 3) * 32] = r3;
        }
    }
    #undef ASYNC_STAGE
    #undef ASYNC_CNORM
    #undef WAIT_ASYNC

    // ---- out = input - final residual (== sum of quantized stages) -------
    float* orow = out + (size_t)row * DDIM;
    #pragma unroll
    for (int dc = 0; dc < 8; ++dc) {
        const int d0 = dc * 32 + half * 8;
        const float4 x0 = *(const float4*)(xrow + d0);
        const float4 x1 = *(const float4*)(xrow + d0 + 4);
        const float4 x2 = *(const float4*)(xrow + d0 + 16);
        const float4 x3 = *(const float4*)(xrow + d0 + 20);
        const float4 r0 = myr[(4 * dc + 0) * 32];
        const float4 r1 = myr[(4 * dc + 1) * 32];
        const float4 r2 = myr[(4 * dc + 2) * 32];
        const float4 r3 = myr[(4 * dc + 3) * 32];
        *(float4*)(orow + d0)      = make_float4(x0.x - r0.x, x0.y - r0.y, x0.z - r0.z, x0.w - r0.w);
        *(float4*)(orow + d0 + 4)  = make_float4(x1.x - r1.x, x1.y - r1.y, x1.z - r1.z, x1.w - r1.w);
        *(float4*)(orow + d0 + 16) = make_float4(x2.x - r2.x, x2.y - r2.y, x2.z - r2.z, x2.w - r2.w);
        *(float4*)(orow + d0 + 20) = make_float4(x3.x - r3.x, x3.y - r3.y, x3.z - r3.z, x3.w - r3.w);
    }
}

// ---------------------------------------------------------------------------
extern "C" void kernel_launch(void* const* d_in, const int* in_sizes, int n_in,
                              void* d_out, int out_size, void* d_ws, size_t ws_size,
                              hipStream_t stream) {
    (void)in_sizes; (void)n_in; (void)out_size; (void)ws_size;
    const float* x  = (const float*)d_in[0];   // [N, D] f32
    const float* cb = (const float*)d_in[1];   // [S, K, D] f32
    float* out = (float*)d_out;                // [N, D] f32

    // Workspace layout: bf16 codebooks (3 MB) then codeword norms (24 KB).
    __bf16* cb16 = (__bf16*)d_ws;
    float*  cnorm = (float*)((char*)d_ws + (size_t)NSTAGE * KCB * DDIM * sizeof(__bf16));

    rvq_prep<<<NSTAGE * KCB, 64, 0, stream>>>(cb, cb16, cnorm);
    rvq_main<<<NROWS / 128, 256, 0, stream>>>(x, cb, cb16, cnorm, out);
}
